// Qwen2MoEMLPLayer_22162031247681
// MI455X (gfx1250) — matrix operand order repaired and checked by reading, not compiled, not run on hardware
//
#include <hip/hip_runtime.h>
#include <hip/hip_bf16.h>
#include <math.h>

// Qwen2 MoE MLP: S=2048,B=1 -> T=2048 tokens, top-2 of 8 experts, H=1024, F=1408
#define T_TOK 2048
#define H_DIM 1024
#define E_NUM 8
#define K_TOP 2
#define F_DIM 1408
#define TK (T_TOK * K_TOP)          // 4096 token-expert rows total
#define ROWS_PAD (TK + 64)          // padded scratch rows (tile overrun guard)

typedef __attribute__((ext_vector_type(16))) __bf16 v16bf;
typedef __attribute__((ext_vector_type(8)))  float  v8f;
typedef unsigned short ushort_t;
typedef unsigned int   uint_t;

// ---- CDNA5 async global->LDS copy (ASYNCcnt path), guarded so compile degrades gracefully ----
#if defined(__has_builtin)
#  if __has_builtin(__builtin_amdgcn_global_load_async_to_lds_b128)
#    define USE_ASYNC_LDS 1
#  endif
#endif
#ifndef USE_ASYNC_LDS
#  define USE_ASYNC_LDS 0
#endif

#if USE_ASYNC_LDS
typedef int v4i_t __attribute__((vector_size(16)));
typedef __attribute__((address_space(1))) v4i_t* gp128_t;   // global (AS1) 16B vector ptr
typedef __attribute__((address_space(3))) v4i_t* lp128_t;   // LDS (AS3) 16B vector ptr
#endif

__device__ __forceinline__ void cp16_g2lds(const ushort_t* __restrict__ g, ushort_t* l) {
#if USE_ASYNC_LDS
  __builtin_amdgcn_global_load_async_to_lds_b128((gp128_t)g, (lp128_t)l, 0, 0);
#else
  *(uint4*)l = *(const uint4*)g;
#endif
}
__device__ __forceinline__ void wait_async_lds() {
#if USE_ASYNC_LDS
  asm volatile("s_wait_asynccnt 0x0" ::: "memory");
#endif
}

__device__ __forceinline__ ushort_t f2bf(float f) {
  uint_t u = __float_as_uint(f);
  uint_t r = u + 0x7FFFu + ((u >> 16) & 1u);   // round-to-nearest-even
  return (ushort_t)(r >> 16);
}

// ---------------- stage 0: zero expert counters ----------------
__global__ void moe_zero_counts(int* counts) {
  if (threadIdx.x < E_NUM) counts[threadIdx.x] = 0;
}

// ---------------- stage 1a: activations fp32 -> bf16 ----------------
__global__ void moe_cvt_bf16(const float* __restrict__ x, ushort_t* __restrict__ xb, int n) {
  int i = (blockIdx.x * blockDim.x + threadIdx.x) * 4;
#pragma unroll
  for (int j = 0; j < 4; ++j)
    if (i + j < n) xb[i + j] = f2bf(x[i + j]);
}

// ---------------- stage 1b: weights fp32 [R][C] -> bf16 transposed [C][R], per expert ----------
__global__ void moe_transpose_cvt(const float* __restrict__ W, ushort_t* __restrict__ Wt,
                                  int R, int C) {
  __shared__ ushort_t tile[32][33];
  const size_t eoff = (size_t)blockIdx.z * (size_t)R * C;
  const float* We = W + eoff;
  ushort_t* Wte = Wt + eoff;
  const int c0 = blockIdx.x * 32, r0 = blockIdx.y * 32;
  const int tc = threadIdx.x & 31, t4 = (threadIdx.x >> 5) * 4;
#pragma unroll
  for (int j = 0; j < 4; ++j)   // coalesced along C
    tile[t4 + j][tc] = f2bf(We[(size_t)(r0 + t4 + j) * C + c0 + tc]);
  __syncthreads();
#pragma unroll
  for (int j = 0; j < 4; ++j)   // coalesced along R
    Wte[(size_t)(c0 + t4 + j) * R + r0 + tc] = tile[tc][t4 + j];
}

// ---------------- stage 2: router (one wave32 per token) ----------------
__global__ void moe_router(const float* __restrict__ x, const float* __restrict__ rw,
                           int* __restrict__ counts, int* __restrict__ eids,
                           int* __restrict__ ranks, float* __restrict__ pvals) {
  const int wave = threadIdx.x >> 5;
  const int lane = threadIdx.x & 31;
  const int t = blockIdx.x * 8 + wave;
  if (t >= T_TOK) return;
  float acc[E_NUM];
#pragma unroll
  for (int e = 0; e < E_NUM; ++e) acc[e] = 0.f;
  const float* xr = x + (size_t)t * H_DIM;
  for (int h = lane; h < H_DIM; h += 32) {
    float xv = xr[h];
    const float* rr = rw + (size_t)h * E_NUM;
#pragma unroll
    for (int e = 0; e < E_NUM; ++e) acc[e] += xv * rr[e];
  }
#pragma unroll
  for (int e = 0; e < E_NUM; ++e) {
#pragma unroll
    for (int off = 16; off > 0; off >>= 1)
      acc[e] += __shfl_xor(acc[e], off, 32);
  }
  if (lane == 0) {
    float m = acc[0];
#pragma unroll
    for (int e = 1; e < E_NUM; ++e) m = fmaxf(m, acc[e]);
    float p[E_NUM], s = 0.f;
#pragma unroll
    for (int e = 0; e < E_NUM; ++e) { p[e] = __expf(acc[e] - m); s += p[e]; }
    float inv = 1.f / s;
#pragma unroll
    for (int e = 0; e < E_NUM; ++e) p[e] *= inv;
    int e0 = 0;
#pragma unroll
    for (int e = 1; e < E_NUM; ++e) if (p[e] > p[e0]) e0 = e;   // ties -> lower idx (matches top_k)
    int e1 = (e0 == 0) ? 1 : 0;
#pragma unroll
    for (int e = 0; e < E_NUM; ++e) if (e != e0 && p[e] > p[e1]) e1 = e;
    int r0 = atomicAdd(&counts[e0], 1);
    int r1 = atomicAdd(&counts[e1], 1);
    eids[t * 2 + 0] = e0;  eids[t * 2 + 1] = e1;
    ranks[t * 2 + 0] = r0; ranks[t * 2 + 1] = r1;
    pvals[t * 2 + 0] = p[e0]; pvals[t * 2 + 1] = p[e1];
  }
}

// ---------------- stage 3: exclusive prefix over E=8 counts ----------------
__global__ void moe_offsets(const int* __restrict__ counts, int* __restrict__ offsets) {
  if (threadIdx.x == 0) {
    int s = 0;
    for (int e = 0; e < E_NUM; ++e) { offsets[e] = s; s += counts[e]; }
  }
}

// ---------------- stage 4: build slot maps ----------------
__global__ void moe_scatter(const int* __restrict__ eids, const int* __restrict__ ranks,
                            const int* __restrict__ offsets,
                            int* __restrict__ slots, int* __restrict__ toks) {
  int i = blockIdx.x * blockDim.x + threadIdx.x;
  if (i >= TK) return;
  int e = eids[i];
  int slot = offsets[e] + ranks[i];
  slots[i] = slot;        // token-k -> global compact row
  toks[slot] = i >> 1;    // global compact row -> token id
}

// ---------------- stage 5: grouped fc1 + SwiGLU ----------------
// 64x64 C-tile per 256-thread block; 8 wave32s in 2(M)x4(N); 4 WMMA / wave / K-step.
__launch_bounds__(256)
__global__ void moe_fc1_swiglu(const ushort_t* __restrict__ xb, const ushort_t* __restrict__ w1t,
                               const int* __restrict__ counts, const int* __restrict__ offsets,
                               const int* __restrict__ toks, ushort_t* __restrict__ G) {
  const int e = blockIdx.z;
  const int cnt = counts[e];
  const int row0 = blockIdx.y * 64;
  if (row0 >= cnt) return;
  const int base = offsets[e];
  const int n0 = blockIdx.x * 64;

  __shared__ ushort_t As[64][56];   // [M][K] bf16; 112B rows: b128-aligned, conflict-free frags
  __shared__ ushort_t Bg[64][56];   // [N][K]
  __shared__ ushort_t Bu[64][56];
  __shared__ int strow[64];

  const int tid = threadIdx.x;
  if (tid < 64)  // clamp to token 0: garbage rows only feed store-guarded C rows
    strow[tid] = (row0 + tid < cnt) ? toks[base + row0 + tid] : 0;
  __syncthreads();

  const int lane = tid & 31;
  const int wv = tid >> 5, wm = wv >> 2, wn = wv & 3;
  const int half = lane >> 4, r = lane & 15;

  v8f c00 = {}, c01 = {}, c10 = {}, c11 = {};   // [mfrag][gate|up]

  const ushort_t* wg = w1t + (size_t)e * (2 * F_DIM) * H_DIM + (size_t)n0 * H_DIM; // N-major bf16
  const ushort_t* wu = wg + (size_t)F_DIM * H_DIM;

  const int srow = tid >> 2;            // 0..63
  const int skq  = (tid & 3) * 8;       // 8 bf16 = 16B per thread
  const ushort_t* xrow = xb + (size_t)strow[srow] * H_DIM;

  union Frag { v16bf v; uint_t u[8]; };
  Frag a0, a1, bg, bu;

  for (int kb = 0; kb < H_DIM; kb += 32) {
    cp16_g2lds(xrow + kb + skq,                         &As[srow][skq]);
    cp16_g2lds(wg + (size_t)srow * H_DIM + kb + skq,    &Bg[srow][skq]);
    cp16_g2lds(wu + (size_t)srow * H_DIM + kb + skq,    &Bu[srow][skq]);
    wait_async_lds();
    __syncthreads();

    const int am0 = wm * 32 + r, am1 = am0 + 16;
#pragma unroll
    for (int i = 0; i < 8; ++i) {        // ISA 16-bit A layout: K pairs, half h takes K 8h..8h+7 and 16+8h..16+8h+7
      int k0 = ((i < 4) ? 0 : 16) + half * 8 + 2 * (i & 3);
      a0.u[i] = *(const uint_t*)&As[am0][k0];
      a1.u[i] = *(const uint_t*)&As[am1][k0];
    }
    const int bn = wn * 16 + r;
#pragma unroll
    for (int j = 0; j < 8; ++j) {        // B: column per lane, K pairs
      int k0 = ((j < 4) ? 0 : 16) + half * 8 + 2 * (j & 3);
      bg.u[j] = *(const uint_t*)&Bg[bn][k0];
      bu.u[j] = *(const uint_t*)&Bu[bn][k0];
    }
    c00 = __builtin_amdgcn_wmma_f32_16x16x32_bf16(false, a0.v, false, bg.v, (short)0, c00, false, false);
    c01 = __builtin_amdgcn_wmma_f32_16x16x32_bf16(false, a0.v, false, bu.v, (short)0, c01, false, false);
    c10 = __builtin_amdgcn_wmma_f32_16x16x32_bf16(false, a1.v, false, bg.v, (short)0, c10, false, false);
    c11 = __builtin_amdgcn_wmma_f32_16x16x32_bf16(false, a1.v, false, bu.v, (short)0, c11, false, false);
    __syncthreads();
  }

  // epilogue: g = silu(gate) * up -> bf16 G
#pragma unroll
  for (int j = 0; j < 8; ++j) {
    const int col = n0 + wn * 16 + r;
    int mA = wm * 32 + half * 8 + j;
    if (row0 + mA < cnt) {
      float g0 = c00[j], u0 = c01[j];
      G[(size_t)(base + row0 + mA) * F_DIM + col] = f2bf(g0 / (1.f + __expf(-g0)) * u0);
    }
    int mB = mA + 16;
    if (row0 + mB < cnt) {
      float g1 = c10[j], u1 = c11[j];
      G[(size_t)(base + row0 + mB) * F_DIM + col] = f2bf(g1 / (1.f + __expf(-g1)) * u1);
    }
  }
}

// ---------------- stage 6: grouped fc2 ----------------
__launch_bounds__(256)
__global__ void moe_fc2(const ushort_t* __restrict__ G, const ushort_t* __restrict__ w2t,
                        const int* __restrict__ counts, const int* __restrict__ offsets,
                        float* __restrict__ O) {
  const int e = blockIdx.z;
  const int cnt = counts[e];
  const int row0 = blockIdx.y * 64;
  if (row0 >= cnt) return;
  const int base = offsets[e];
  const int n0 = blockIdx.x * 64;

  __shared__ ushort_t As[64][56];
  __shared__ ushort_t Bs[64][56];

  const int tid = threadIdx.x;
  const int lane = tid & 31;
  const int wv = tid >> 5, wm = wv >> 2, wn = wv & 3;
  const int half = lane >> 4, r = lane & 15;

  v8f c0 = {}, c1 = {};
  const ushort_t* wb = w2t + (size_t)e * H_DIM * F_DIM + (size_t)n0 * F_DIM;  // N-major bf16

  const int srow = tid >> 2;
  const int skq  = (tid & 3) * 8;
  const ushort_t* ga = G + (size_t)(base + row0 + srow) * F_DIM;  // padded rows absorb overrun

  union Frag { v16bf v; uint_t u[8]; };
  Frag a0, a1, b;

  for (int kb = 0; kb < F_DIM; kb += 32) {
    cp16_g2lds(ga + kb + skq,                          &As[srow][skq]);
    cp16_g2lds(wb + (size_t)srow * F_DIM + kb + skq,   &Bs[srow][skq]);
    wait_async_lds();
    __syncthreads();

    const int am0 = wm * 32 + r, am1 = am0 + 16;
#pragma unroll
    for (int i = 0; i < 8; ++i) {
      int k0 = ((i < 4) ? 0 : 16) + half * 8 + 2 * (i & 3);
      a0.u[i] = *(const uint_t*)&As[am0][k0];
      a1.u[i] = *(const uint_t*)&As[am1][k0];
    }
    const int bn = wn * 16 + r;
#pragma unroll
    for (int j = 0; j < 8; ++j) {
      int k0 = ((j < 4) ? 0 : 16) + half * 8 + 2 * (j & 3);
      b.u[j] = *(const uint_t*)&Bs[bn][k0];
    }
    c0 = __builtin_amdgcn_wmma_f32_16x16x32_bf16(false, a0.v, false, b.v, (short)0, c0, false, false);
    c1 = __builtin_amdgcn_wmma_f32_16x16x32_bf16(false, a1.v, false, b.v, (short)0, c1, false, false);
    __syncthreads();
  }
#pragma unroll
  for (int j = 0; j < 8; ++j) {
    const int col = n0 + wn * 16 + r;
    int mA = wm * 32 + half * 8 + j;
    if (row0 + mA < cnt) O[(size_t)(base + row0 + mA) * H_DIM + col] = c0[j];
    int mB = mA + 16;
    if (row0 + mB < cnt) O[(size_t)(base + row0 + mB) * H_DIM + col] = c1[j];
  }
}

// ---------------- stage 7: prob-weighted combine (deterministic, no atomics) ----------------
__global__ void moe_combine(const float* __restrict__ O, const int* __restrict__ slots,
                            const float* __restrict__ pvals, float* __restrict__ out) {
  int t = blockIdx.x;
  int s0 = slots[t * 2 + 0], s1 = slots[t * 2 + 1];
  float p0 = pvals[t * 2 + 0], p1 = pvals[t * 2 + 1];
  const float* o0 = O + (size_t)s0 * H_DIM;
  const float* o1 = O + (size_t)s1 * H_DIM;
  float* op = out + (size_t)t * H_DIM;
  for (int h = threadIdx.x; h < H_DIM; h += blockDim.x)
    op[h] = p0 * o0[h] + p1 * o1[h];
}

extern "C" void kernel_launch(void* const* d_in, const int* in_sizes, int n_in,
                              void* d_out, int out_size, void* d_ws, size_t ws_size,
                              hipStream_t stream) {
  const float* x  = (const float*)d_in[0];   // [T, H]
  const float* rw = (const float*)d_in[1];   // [H, E]
  const float* w1 = (const float*)d_in[2];   // [E, H, 2F]
  const float* w2 = (const float*)d_in[3];   // [E, F, H]
  float* out = (float*)d_out;                // [T, H]

  char* ws = (char*)d_ws;
  size_t off = 0;
  auto carve = [&](size_t bytes) -> void* {
    void* p = ws + off;
    off = (off + bytes + 255) & ~(size_t)255;
    return p;
  };
  int*      counts  = (int*)carve(E_NUM * sizeof(int));
  int*      offsets = (int*)carve(E_NUM * sizeof(int));
  int*      eids    = (int*)carve(TK * sizeof(int));
  int*      ranks   = (int*)carve(TK * sizeof(int));
  float*    pvals   = (float*)carve(TK * sizeof(float));
  int*      slots   = (int*)carve(TK * sizeof(int));
  int*      toks    = (int*)carve(ROWS_PAD * sizeof(int));
  ushort_t* xb      = (ushort_t*)carve((size_t)T_TOK * H_DIM * sizeof(ushort_t));
  ushort_t* w1t     = (ushort_t*)carve((size_t)E_NUM * 2 * F_DIM * H_DIM * sizeof(ushort_t));
  ushort_t* w2t     = (ushort_t*)carve((size_t)E_NUM * F_DIM * H_DIM * sizeof(ushort_t));
  ushort_t* G       = (ushort_t*)carve((size_t)ROWS_PAD * F_DIM * sizeof(ushort_t));
  float*    O       = (float*)carve((size_t)ROWS_PAD * H_DIM * sizeof(float));

  moe_zero_counts<<<1, 32, 0, stream>>>(counts);
  moe_cvt_bf16<<<(T_TOK * H_DIM / 4 + 255) / 256, 256, 0, stream>>>(x, xb, T_TOK * H_DIM);
  moe_transpose_cvt<<<dim3(2 * F_DIM / 32, H_DIM / 32, E_NUM), 256, 0, stream>>>(w1, w1t, H_DIM, 2 * F_DIM);
  moe_transpose_cvt<<<dim3(H_DIM / 32, F_DIM / 32, E_NUM), 256, 0, stream>>>(w2, w2t, F_DIM, H_DIM);
  moe_router<<<T_TOK / 8, 256, 0, stream>>>(x, rw, counts, eids, ranks, pvals);
  moe_offsets<<<1, 32, 0, stream>>>(counts, offsets);
  moe_scatter<<<(TK + 255) / 256, 256, 0, stream>>>(eids, ranks, offsets, slots, toks);
  moe_fc1_swiglu<<<dim3(F_DIM / 64, T_TOK / 64, E_NUM), 256, 0, stream>>>(xb, w1t, counts, offsets, toks, G);
  moe_fc2<<<dim3(H_DIM / 64, T_TOK / 64, E_NUM), 256, 0, stream>>>(G, w2t, counts, offsets, O);
  moe_combine<<<T_TOK, 256, 0, stream>>>(O, slots, pvals, out);
}
